// ContextAttentionAdapterWrapper_36936718745757
// MI455X (gfx1250) — matrix operand order repaired and checked by reading, not compiled, not run on hardware
//
#include <hip/hip_runtime.h>
#include <hip/hip_bf16.h>
#include <math.h>

// ---------- types for WMMA fragments (wave32, 16x16x32 f16 -> f32) ----------
typedef __attribute__((ext_vector_type(16))) _Float16 v16h;
typedef __attribute__((ext_vector_type(8)))  _Float16 half8;
typedef __attribute__((ext_vector_type(8)))  float    v8f;

// dims
#define N_SEQ   896
#define DH      3072
#define NCTX    64
#define JLEN    127
#define DC      1024
#define INNER   512
#define HEADS   8
#define DHEAD   64
#define JP      128   // j + null token

__device__ inline v8f zero8() { v8f z = {0.f,0.f,0.f,0.f,0.f,0.f,0.f,0.f}; return z; }

// A fragment: 16x32 f16, source row-major, base points at [row0][k0], ld in elements.
// lane l (g=l>>4, m=l&15): elements 0..7 = K = 8g..8g+7 ; 8..15 = K = 16+8g..16+8g+7
__device__ inline v16h load_a(const _Float16* base, int ld, int lane) {
    int g = lane >> 4, m = lane & 15;
    half8 lo = *(const half8*)(base + (size_t)m * ld + 8 * g);
    half8 hi = *(const half8*)(base + (size_t)m * ld + 8 * g + 16);
    return __builtin_shufflevector(lo, hi, 0,1,2,3,4,5,6,7,8,9,10,11,12,13,14,15);
}

// B fragment: 32x16 f16, source stored transposed: BT[n][k], k contiguous, base at [n0][k0].
// lane l (g=l>>4, n=l&15): elements 0..7 = K = 8g..8g+7 ; 8..15 = K = 16+8g..16+8g+7
__device__ inline v16h load_b(const _Float16* base, int ld, int lane) {
    int g = lane >> 4, n = lane & 15;
    const half8* p = (const half8*)(base + (size_t)n * ld + 8 * g);
    half8 lo = p[0];
    half8 hi = p[2];
    return __builtin_shufflevector(lo, hi, 0,1,2,3,4,5,6,7,8,9,10,11,12,13,14,15);
}

__device__ inline v8f wmma16(v16h a, v16h b, v8f c) {
    return __builtin_amdgcn_wmma_f32_16x16x32_f16(false, a, false, b, (short)0, c, false, false);
}

// 256-thread block sum reduction (wave32)
__device__ inline float block_reduce_sum(float v, float* sbuf) {
    __syncthreads();
    int tid = threadIdx.x;
    #pragma unroll
    for (int off = 16; off >= 1; off >>= 1) v += __shfl_xor(v, off, 32);
    if ((tid & 31) == 0) sbuf[tid >> 5] = v;
    __syncthreads();
    float r = (tid < 8) ? sbuf[tid] : 0.f;
    if ((tid >> 5) == 0) {
        #pragma unroll
        for (int off = 4; off >= 1; off >>= 1) r += __shfl_xor(r, off, 32);
        if (tid == 0) sbuf[0] = r;
    }
    __syncthreads();
    return sbuf[0];
}

// ---------- K0: rank-1 folds: ep[n]=emb[n]·Wp ; wop[i]=Wo[i,:]·Wp ; bobp=bo·Wp+bp ----------
__global__ void k_prep(const float* __restrict__ emb, const float* __restrict__ Wo,
                       const float* __restrict__ bo, const float* __restrict__ Wp,
                       const float* __restrict__ bp,
                       float* ep, float* wop, float* bobp) {
    __shared__ float sbuf[32];
    int b = blockIdx.x;
    float partial = 0.f;
    if (b < N_SEQ) {
        const float* row = emb + (size_t)b * DH;
        for (int k = threadIdx.x; k < DH; k += 256) partial += row[k] * Wp[k];
        float s = block_reduce_sum(partial, sbuf);
        if (threadIdx.x == 0) ep[b] = s;
    } else if (b < N_SEQ + INNER) {
        int i = b - N_SEQ;
        const float* row = Wo + (size_t)i * DH;
        for (int k = threadIdx.x; k < DH; k += 256) partial += row[k] * Wp[k];
        float s = block_reduce_sum(partial, sbuf);
        if (threadIdx.x == 0) wop[i] = s;
    } else {
        for (int k = threadIdx.x; k < DH; k += 256) partial += bo[k] * Wp[k];
        float s = block_reduce_sum(partial, sbuf);
        if (threadIdx.x == 0) bobp[0] = s + bp[0];
    }
}

// ---------- transpose+convert fp32 weight [K][Ntot] col range [c0,c0+N) -> f16 [N][K] ----------
__global__ void k_transpose16(const float* __restrict__ src, _Float16* __restrict__ dst,
                              int K, int Ntot, int c0, int N) {
    size_t total = (size_t)N * K;
    for (size_t idx = (size_t)blockIdx.x * blockDim.x + threadIdx.x; idx < total;
         idx += (size_t)gridDim.x * blockDim.x) {
        size_t n = idx / K, k = idx - n * K;
        dst[idx] = (_Float16)src[k * Ntot + c0 + n];
    }
}

// ---------- fill null_k as j=0 row of every context's K buffer ----------
__global__ void k_fill_null(const float* __restrict__ null_k, _Float16* __restrict__ Kb) {
    int idx = blockIdx.x * blockDim.x + threadIdx.x;
    if (idx < NCTX * INNER) {
        int c = idx >> 9, t = idx & 511;
        Kb[(size_t)c * JP * INNER + t] = (_Float16)null_k[t];
    }
}

// ---------- wv[k][h] = sum_d Wkv[k][512+h*64+d] * wop[h*64+d] ----------
__global__ void k_wv(const float* __restrict__ Wkv, const float* __restrict__ wop,
                     float* __restrict__ wv) {
    int idx = blockIdx.x * blockDim.x + threadIdx.x;
    if (idx < DC * HEADS) {
        int k = idx >> 3, h = idx & 7;
        float s = 0.f;
        const float* wr = Wkv + (size_t)k * (2 * INNER) + INNER + h * DHEAD;
        const float* wo = wop + h * DHEAD;
        #pragma unroll 8
        for (int d = 0; d < DHEAD; ++d) s += wr[d] * wo[d];
        wv[idx] = s;
    }
}

// ---------- vw[c][h][0] = null_v_h · wop_h ----------
__global__ void k_vwnull(const float* __restrict__ null_v, const float* __restrict__ wop,
                         float* __restrict__ vwbuf) {
    int idx = blockIdx.x * blockDim.x + threadIdx.x;
    if (idx < NCTX * HEADS) {
        int c = idx >> 3, h = idx & 7;
        float s = 0.f;
        #pragma unroll 8
        for (int d = 0; d < DHEAD; ++d) s += null_v[h * DHEAD + d] * wop[h * DHEAD + d];
        vwbuf[(size_t)c * (HEADS * JP) + h * JP + 0] = s;
    }
}

// ---------- layernorm row -> f16 ----------
__global__ void k_ln(const float* __restrict__ x, const float* __restrict__ g,
                     const float* __restrict__ be, _Float16* __restrict__ y, int cols) {
    __shared__ float sbuf[32];
    size_t r = blockIdx.x;
    const float* row = x + r * cols;
    float s = 0.f, s2 = 0.f;
    for (int k = threadIdx.x; k < cols; k += 256) { float v = row[k]; s += v; s2 += v * v; }
    s  = block_reduce_sum(s,  sbuf);
    s2 = block_reduce_sum(s2, sbuf);
    float mean = s / cols;
    float var  = s2 / cols - mean * mean;
    float rstd = rsqrtf(var + 1e-5f);
    _Float16* yr = y + r * cols;
    for (int k = threadIdx.x; k < cols; k += 256)
        yr[k] = (_Float16)((row[k] - mean) * rstd * g[k] + be[k]);
}

// ---------- software-pipelined wave GEMM body: 16xK @ Kx64 -> acc[4] ----------
// A row-major (ld = K), BT transposed (ld = K). Double-buffered fragments so the
// s_wait before each WMMA group covers loads issued one full k-step earlier.
template <int K>
__device__ inline void gemm_tile_16x64(const _Float16* __restrict__ Arow,
                                       const _Float16* __restrict__ Bbase,
                                       int lane, v8f acc[4]) {
    v16h a_cur = load_a(Arow, K, lane);
    v16h b_cur0 = load_b(Bbase + (size_t)0  * K, K, lane);
    v16h b_cur1 = load_b(Bbase + (size_t)16 * K, K, lane);
    v16h b_cur2 = load_b(Bbase + (size_t)32 * K, K, lane);
    v16h b_cur3 = load_b(Bbase + (size_t)48 * K, K, lane);
    for (int k0 = 0; k0 < K - 32; k0 += 32) {
        int kn = k0 + 32;
        __builtin_prefetch(Arow + kn + 32, 0, 1);
        v16h a_nxt  = load_a(Arow + kn, K, lane);
        v16h b_nxt0 = load_b(Bbase + (size_t)0  * K + kn, K, lane);
        v16h b_nxt1 = load_b(Bbase + (size_t)16 * K + kn, K, lane);
        v16h b_nxt2 = load_b(Bbase + (size_t)32 * K + kn, K, lane);
        v16h b_nxt3 = load_b(Bbase + (size_t)48 * K + kn, K, lane);
        acc[0] = wmma16(a_cur, b_cur0, acc[0]);
        acc[1] = wmma16(a_cur, b_cur1, acc[1]);
        acc[2] = wmma16(a_cur, b_cur2, acc[2]);
        acc[3] = wmma16(a_cur, b_cur3, acc[3]);
        a_cur = a_nxt;
        b_cur0 = b_nxt0; b_cur1 = b_nxt1; b_cur2 = b_nxt2; b_cur3 = b_nxt3;
    }
    acc[0] = wmma16(a_cur, b_cur0, acc[0]);
    acc[1] = wmma16(a_cur, b_cur1, acc[1]);
    acc[2] = wmma16(a_cur, b_cur2, acc[2]);
    acc[3] = wmma16(a_cur, b_cur3, acc[3]);
}

// ---------- Q = lnE(896x3072) @ WqT  -> f16 (896x512). wave: 16x64 tile ----------
__global__ void k_gemm_q(const _Float16* __restrict__ A, const _Float16* __restrict__ BT,
                         _Float16* __restrict__ Q) {
    int lane = threadIdx.x & 31, w = threadIdx.x >> 5;
    int mt = blockIdx.x * 4 + w;     // 0..55
    int n0 = blockIdx.y * 64;        // 0..448
    v8f acc[4];
    #pragma unroll
    for (int t = 0; t < 4; ++t) acc[t] = zero8();
    gemm_tile_16x64<DH>(A + (size_t)(mt * 16) * DH, BT + (size_t)n0 * DH, lane, acc);
    int g = lane >> 4, nn = lane & 15;
    #pragma unroll
    for (int t = 0; t < 4; ++t)
        #pragma unroll
        for (int v = 0; v < 8; ++v) {
            int row = mt * 16 + 8 * g + v;
            int col = n0 + t * 16 + nn;
            Q[(size_t)row * INNER + col] = (_Float16)acc[t][v];
        }
}

// ---------- K = lnC(8128x1024) @ WkT -> f16 into Kb[c][1+jj][512] (row remap) ----------
__global__ void k_gemm_k(const _Float16* __restrict__ A, const _Float16* __restrict__ BT,
                         _Float16* __restrict__ Kb) {
    int lane = threadIdx.x & 31, w = threadIdx.x >> 5;
    int mt = blockIdx.x * 4 + w;     // 0..507
    int n0 = blockIdx.y * 64;
    v8f acc[4];
    #pragma unroll
    for (int t = 0; t < 4; ++t) acc[t] = zero8();
    gemm_tile_16x64<DC>(A + (size_t)(mt * 16) * DC, BT + (size_t)n0 * DC, lane, acc);
    int g = lane >> 4, nn = lane & 15;
    #pragma unroll
    for (int t = 0; t < 4; ++t)
        #pragma unroll
        for (int v = 0; v < 8; ++v) {
            int r = mt * 16 + 8 * g + v;          // 0..8127
            int cc = r / JLEN, jj = r - cc * JLEN;
            int col = n0 + t * 16 + nn;
            Kb[((size_t)cc * JP + 1 + jj) * INNER + col] = (_Float16)acc[t][v];
        }
}

// ---------- vw[c][h][1+jj] = lnC row · wv[:,h] ----------
__global__ void k_vw_rows(const _Float16* __restrict__ lnC, const float* __restrict__ wv,
                          float* __restrict__ vwbuf) {
    __shared__ float srow[DC];
    __shared__ float sbuf[32];
    int r = blockIdx.x;
    int c = r / JLEN, jj = r - c * JLEN;
    const _Float16* row = lnC + (size_t)r * DC;
    for (int k = threadIdx.x; k < DC; k += 256) srow[k] = (float)row[k];
    __syncthreads();
    for (int h = 0; h < HEADS; ++h) {
        float p = 0.f;
        for (int k = threadIdx.x; k < DC; k += 256) p += srow[k] * wv[k * HEADS + h];
        p = block_reduce_sum(p, sbuf);
        if (threadIdx.x == 0) vwbuf[(size_t)c * (HEADS * JP) + h * JP + 1 + jj] = p;
    }
}

// ---------- attention: sim = QK^T (WMMA), masked softmax, fold with vw, softplus ----------
__global__ void k_attn(const _Float16* __restrict__ Q, const _Float16* __restrict__ Kb,
                       const float* __restrict__ vwbuf, const unsigned char* __restrict__ mask,
                       const float* __restrict__ ep, const float* __restrict__ bobp,
                       float* __restrict__ out) {
    int lane = threadIdx.x & 31, w = threadIdx.x >> 5;
    int c = blockIdx.x;
    int ibase = blockIdx.y * 64 + w * 16;
    int g = lane >> 4, nn = lane & 15;
    const float NEG = -3.0e38f;
    const float scale = 0.125f;   // 64^-0.5

    unsigned mbits = 0;
    #pragma unroll
    for (int jt = 0; jt < 8; ++jt) {
        int j = jt * 16 + nn;
        bool ok = (j == 0) ? true : (mask[j - 1] != 0);
        if (ok) mbits |= (1u << jt);
    }

    float rowres[8];
    #pragma unroll
    for (int v = 0; v < 8; ++v) rowres[v] = 0.f;

    for (int h = 0; h < HEADS; ++h) {
        const _Float16* Qbase = Q + (size_t)ibase * INNER + h * DHEAD;
        const _Float16* Kh    = Kb + (size_t)c * JP * INNER + h * DHEAD;
        v16h a0 = load_a(Qbase,      INNER, lane);
        v16h a1 = load_a(Qbase + 32, INNER, lane);
        v8f s[8];
        // double-buffered j-tile loop (fully unrolled; guard folds away)
        v16h b0c = load_b(Kh,      INNER, lane);
        v16h b1c = load_b(Kh + 32, INNER, lane);
        #pragma unroll
        for (int jt = 0; jt < 8; ++jt) {
            v16h b0n, b1n;
            if (jt < 7) {
                const _Float16* Kn = Kh + (size_t)(jt + 1) * 16 * INNER;
                b0n = load_b(Kn,      INNER, lane);
                b1n = load_b(Kn + 32, INNER, lane);
            }
            s[jt] = zero8();
            s[jt] = wmma16(a0, b0c, s[jt]);
            s[jt] = wmma16(a1, b1c, s[jt]);
            b0c = b0n; b1c = b1n;
        }
        float vwv[8];
        #pragma unroll
        for (int jt = 0; jt < 8; ++jt)
            vwv[jt] = vwbuf[(size_t)c * (HEADS * JP) + h * JP + jt * 16 + nn];

        #pragma unroll
        for (int v = 0; v < 8; ++v) {
            float sv[8]; float rmax = NEG;
            #pragma unroll
            for (int jt = 0; jt < 8; ++jt) {
                float x = ((mbits >> jt) & 1u) ? (float)s[jt][v] * scale : NEG;
                sv[jt] = x;
                rmax = fmaxf(rmax, x);
            }
            #pragma unroll
            for (int off = 8; off >= 1; off >>= 1)
                rmax = fmaxf(rmax, __shfl_xor(rmax, off, 32));
            float se = 0.f, swv = 0.f;
            #pragma unroll
            for (int jt = 0; jt < 8; ++jt) {
                float e = __expf(sv[jt] - rmax);
                se += e; swv += e * vwv[jt];
            }
            #pragma unroll
            for (int off = 8; off >= 1; off >>= 1) {
                se  += __shfl_xor(se,  off, 32);
                swv += __shfl_xor(swv, off, 32);
            }
            rowres[v] += swv / se;
        }
    }
    if (nn == 0) {
        float bb = bobp[0];
        #pragma unroll
        for (int v = 0; v < 8; ++v) {
            int n = ibase + 8 * g + v;
            float x = rowres[v] + ep[n] + bb;
            float sp = (x > 20.f) ? x : log1pf(__expf(x));
            out[(size_t)n * NCTX + c] = sp;  // (b, n, c)
        }
    }
}

extern "C" void kernel_launch(void* const* d_in, const int* in_sizes, int n_in,
                              void* d_out, int out_size, void* d_ws, size_t ws_size,
                              hipStream_t stream) {
    const float* emb    = (const float*)d_in[0];
    const float* ctx    = (const float*)d_in[1];
    const unsigned char* cmask = (const unsigned char*)d_in[2];
    const float* qg     = (const float*)d_in[3];
    const float* qb     = (const float*)d_in[4];
    const float* kvg    = (const float*)d_in[5];
    const float* kvb    = (const float*)d_in[6];
    const float* Wq     = (const float*)d_in[7];
    const float* Wkv    = (const float*)d_in[8];
    const float* null_k = (const float*)d_in[9];
    const float* null_v = (const float*)d_in[10];
    const float* Wo     = (const float*)d_in[11];
    const float* bo     = (const float*)d_in[12];
    const float* Wp     = (const float*)d_in[13];
    const float* bp     = (const float*)d_in[14];
    float* out = (float*)d_out;

    char* ws = (char*)d_ws;
    size_t off = 0;
    auto alloc = [&](size_t bytes) { size_t o = off; off = (off + bytes + 255) & ~(size_t)255; return o; };
    float*    ep    = (float*)   (ws + alloc(N_SEQ * 4));
    float*    wop   = (float*)   (ws + alloc(INNER * 4));
    float*    bobp  = (float*)   (ws + alloc(4));
    float*    wv    = (float*)   (ws + alloc(DC * HEADS * 4));
    float*    vwbuf = (float*)   (ws + alloc((size_t)NCTX * HEADS * JP * 4));
    _Float16* lnE   = (_Float16*)(ws + alloc((size_t)N_SEQ * DH * 2));
    _Float16* lnC   = (_Float16*)(ws + alloc((size_t)NCTX * JLEN * DC * 2));
    _Float16* WqT   = (_Float16*)(ws + alloc((size_t)INNER * DH * 2));
    _Float16* WkT   = (_Float16*)(ws + alloc((size_t)INNER * DC * 2));
    _Float16* Qb    = (_Float16*)(ws + alloc((size_t)N_SEQ * INNER * 2));
    _Float16* Kb    = (_Float16*)(ws + alloc((size_t)NCTX * JP * INNER * 2));

    // K0: rank-1 folds (ep, wop, bobp)
    k_prep<<<N_SEQ + INNER + 1, 256, 0, stream>>>(emb, Wo, bo, Wp, bp, ep, wop, bobp);
    // weight transposes -> f16
    k_transpose16<<<6144, 256, 0, stream>>>(Wq,  WqT, DH, INNER,     0, INNER);
    k_transpose16<<<2048, 256, 0, stream>>>(Wkv, WkT, DC, 2 * INNER, 0, INNER);
    // null k rows, wv = Wkv_v @ wop, vw null entries
    k_fill_null<<<(NCTX * INNER + 255) / 256, 256, 0, stream>>>(null_k, Kb);
    k_wv<<<(DC * HEADS + 255) / 256, 256, 0, stream>>>(Wkv, wop, wv);
    k_vwnull<<<(NCTX * HEADS + 255) / 256, 256, 0, stream>>>(null_v, wop, vwbuf);
    // layernorms -> f16
    k_ln<<<N_SEQ, 256, 0, stream>>>(emb, qg, qb, lnE, DH);
    k_ln<<<NCTX * JLEN, 256, 0, stream>>>(ctx, kvg, kvb, lnC, DC);
    // WMMA GEMMs (software-pipelined)
    k_gemm_q<<<dim3(14, 8), 128, 0, stream>>>(lnE, WqT, Qb);
    k_gemm_k<<<dim3(127, 8), 128, 0, stream>>>(lnC, WkT, Kb);
    // vw rows (folded V path)
    k_vw_rows<<<NCTX * JLEN, 256, 0, stream>>>(lnC, wv, vwbuf);
    // attention + softmax + fold + softplus
    k_attn<<<dim3(NCTX, 14), 128, 0, stream>>>(Qb, Kb, vwbuf, cmask, ep, bobp, out);
}
